// SmoothNetWrapper_80066780332764
// MI455X (gfx1250) — matrix operand order repaired and checked by reading, not compiled, not run on hardware
//
#include <hip/hip_runtime.h>
#include <hip/hip_bf16.h>
#include <stdint.h>

// ---------------------------------------------------------------------------
// Types for WMMA fragments
// ---------------------------------------------------------------------------
typedef __attribute__((ext_vector_type(16))) __bf16        bf16x16;
typedef __attribute__((ext_vector_type(8)))  float         f32x8;
typedef __attribute__((ext_vector_type(4)))  unsigned int  u32x4;

union Frag {
    bf16x16        v;
    __bf16         e[16];
    unsigned short h[16];
    u32x4          q[2];
};

// ---------------------------------------------------------------------------
// Fragment loaders (ISA 7.12.2 layouts, wave32)
// A 16x32 bf16: lanes 0-15 row=lane, K = k0+[0..7] and k0+16+[0..7] (k0=kt*32)
//               lanes 16-31 row=lane-16, shifted by +8.
// B 32x16 bf16: lane holds col n = lane%16; same K layout as A:
//               K = k0+8*half+[0..7] and k0+16+8*half+[0..7].  Since every layer is x @ W^T,
//               row-major W (row n, length K) is exactly B column-major.
// ---------------------------------------------------------------------------
__device__ __forceinline__ Frag load_a(const unsigned short* base, int rowStride, int kt) {
    int lane = threadIdx.x & 31;
    int half = lane >> 4;
    int row  = lane & 15;
    const unsigned short* p = base + row * rowStride + kt * 32 + half * 8;
    Frag f;
    f.q[0] = *(const u32x4*)(p);        // K = k0 .. k0+7
    f.q[1] = *(const u32x4*)(p + 16);   // K = k0+16 .. k0+23
    return f;
}

__device__ __forceinline__ Frag load_b(const unsigned short* W, int K, int kt, int nt) {
    int lane = threadIdx.x & 31;
    int half = lane >> 4;
    int col  = lane & 15;
    const unsigned short* p = W + (nt * 16 + col) * K + kt * 32 + half * 8;
    Frag f;
    f.q[0] = *(const u32x4*)(p);
    f.q[1] = *(const u32x4*)(p + 16);
    return f;
}

// C/D fragment element (VGPR j, lane l) = (row j + 8*(l>=16), col l%16)
template<bool RES, bool ACT>
__device__ __forceinline__ void store_c(f32x8 c, const float* bias, unsigned short* out,
                                        int N, int nt) {
    int lane = threadIdx.x & 31;
    int half = lane >> 4;
    int col  = lane & 15;
    int n    = nt * 16 + col;
    float b  = bias[n];
    __bf16* o = (__bf16*)out;
#pragma unroll
    for (int j = 0; j < 8; ++j) {
        int m   = j + half * 8;
        float v = c[j] + b;
        if (ACT) v = (v >= 0.f) ? v : 0.1f * v;
        if (RES) v += (float)o[m * N + n];
        o[m * N + n] = (__bf16)v;        // native v_cvt f32->bf16 (RNE)
    }
}

// Templated layer: hoist all A fragments into registers, stream B fragments.
template<int K, int N, bool RES, bool ACT>
__device__ __forceinline__ void layer_t(const unsigned short* A, unsigned short* O,
                                        const unsigned short* W, const float* bias) {
    Frag a[K / 32];
#pragma unroll
    for (int kt = 0; kt < K / 32; ++kt) a[kt] = load_a(A, K, kt);
    for (int nt = 0; nt < N / 16; ++nt) {
        f32x8 c = {};
#pragma unroll
        for (int kt = 0; kt < K / 32; ++kt) {
            Frag b = load_b(W, K, kt, nt);
            c = __builtin_amdgcn_wmma_f32_16x16x32_bf16(
                    false, a[kt].v, false, b.v, (short)0, c, false, false);
        }
        store_c<RES, ACT>(c, bias, O, N, nt);
    }
}

// ---------------------------------------------------------------------------
// Prep kernels
// ---------------------------------------------------------------------------
__global__ void cvt_bf16_kernel(const float* __restrict__ src,
                                unsigned short* __restrict__ dst, int n) {
    int i = blockIdx.x * 256 + threadIdx.x;
    if (i < n) {
        __bf16 b = (__bf16)src[i];
        dst[i] = *(unsigned short*)&b;
    }
}

// data_pred (1024,72) axis-angle -> d6T bf16 (144,1024) time-contiguous
__global__ void aa_to_d6_kernel(const float* __restrict__ pred,
                                unsigned short* __restrict__ d6T, int total) {
    int i = blockIdx.x * 256 + threadIdx.x;
    if (i >= total) return;                 // 1024*24
    int f = i / 24, j = i % 24;
    float ax = pred[f * 72 + j * 3 + 0];
    float ay = pred[f * 72 + j * 3 + 1];
    float az = pred[f * 72 + j * 3 + 2];
    float ang = sqrtf(ax * ax + ay * ay + az * az);
    float d   = fmaxf(ang, 1e-8f);
    float x = ax / d, y = ay / d, z = az / d;
    float s = sinf(ang), c = cosf(ang), C = 1.f - c;
    float r[6] = { c + x * x * C,  x * y * C - z * s,  x * z * C + y * s,
                   y * x * C + z * s,  c + y * y * C,  y * z * C - x * s };
#pragma unroll
    for (int k = 0; k < 6; ++k) {
        __bf16 b = (__bf16)r[k];
        d6T[(j * 6 + k) * 1024 + f] = *(unsigned short*)&b;
    }
}

// ---------------------------------------------------------------------------
// Fused SmoothNet kernel: one wave per 16-row fragment (fixed window, 16 chans)
// ---------------------------------------------------------------------------
__global__ __launch_bounds__(64) void smoothnet_main(
    const unsigned short* __restrict__ d6T,
    const unsigned short* __restrict__ w_in,  const float* __restrict__ b_in,
    const unsigned short* __restrict__ w1a,   const float* __restrict__ b1a,
    const unsigned short* __restrict__ w1b,   const float* __restrict__ b1b,
    const unsigned short* __restrict__ w2a,   const float* __restrict__ b2a,
    const unsigned short* __restrict__ w2b,   const float* __restrict__ b2b,
    const unsigned short* __restrict__ w_out, const float* __restrict__ b_out,
    float* __restrict__ seq, int nFrag)
{
    __shared__ unsigned short sh_h[2][16 * 512];   // 2 waves * 16KB
    __shared__ unsigned short sh_r[2][16 * 256];   // 2 waves * 8KB

    int wave = threadIdx.x >> 5;
    int frag = blockIdx.x * 2 + wave;
    if (frag >= nFrag) return;                     // wave-uniform

    int nwin = frag / 9;                           // window index (0..992)
    int c0   = (frag % 9) * 16;                    // channel block base
    unsigned short* H = sh_h[wave];
    unsigned short* R = sh_r[wave];

    int lane = threadIdx.x & 31;
    int half = lane >> 4;
    int row  = lane & 15;

    // ---- layer 1: h = lrelu(X @ w_in^T + b_in), K=32, N=512 -------------
    // A from d6T (row stride 1024, base offset nwin): scalar loads since the
    // time offset breaks 16B alignment.
    Frag a;
    {
        const unsigned short* src = d6T + (c0 + row) * 1024 + nwin + half * 8;
#pragma unroll
        for (int i = 0; i < 8; ++i) { a.h[i] = src[i]; a.h[8 + i] = src[16 + i]; }
    }
    for (int nt = 0; nt < 32; ++nt) {
        Frag b  = load_b(w_in, 32, 0, nt);
        f32x8 c = {};
        c = __builtin_amdgcn_wmma_f32_16x16x32_bf16(
                false, a.v, false, b.v, (short)0, c, false, false);
        store_c<false, true>(c, b_in, H, 512, nt);
    }

    // ---- residual blocks ------------------------------------------------
    layer_t<512, 256, false, true>(H, R, w1a, b1a);   // r  = lrelu(h @ w1a^T + b1a)
    layer_t<256, 512, true,  true>(R, H, w1b, b1b);   // h += lrelu(r @ w1b^T + b1b)
    layer_t<512, 256, false, true>(H, R, w2a, b2a);
    layer_t<256, 512, true,  true>(R, H, w2b, b2b);

    // ---- output layer: y = h @ w_out^T + b_out, K=512, N=32 -------------
    {
        Frag ah[16];
#pragma unroll
        for (int kt = 0; kt < 16; ++kt) ah[kt] = load_a(H, 512, kt);
#pragma unroll
        for (int nt = 0; nt < 2; ++nt) {
            f32x8 c = {};
#pragma unroll
            for (int kt = 0; kt < 16; ++kt) {
                Frag b = load_b(w_out, 512, kt, nt);
                c = __builtin_amdgcn_wmma_f32_16x16x32_bf16(
                        false, ah[kt].v, false, b.v, (short)0, c, false, false);
            }
            int   t  = nt * 16 + (lane & 15);       // time within window
            float bo = b_out[t];
#pragma unroll
            for (int j = 0; j < 8; ++j) {
                int m = j + half * 8;               // channel offset
                atomicAdd(&seq[(nwin + t) * 144 + c0 + m], c[j] + bo);
            }
        }
    }
}

// ---------------------------------------------------------------------------
// Finalize: average (analytic counts), 6D -> rotation matrix -> axis-angle
// ---------------------------------------------------------------------------
__global__ void finalize_kernel(const float* __restrict__ seq,
                                float* __restrict__ out, int total) {
    int i = blockIdx.x * 256 + threadIdx.x;
    if (i >= total) return;                 // 1024*24
    int p = i / 24, j = i % 24;
    int lo = (p - 31 > 0) ? (p - 31) : 0;
    int hi = (p < 992) ? p : 992;
    float inv = 1.0f / (float)(hi - lo + 1);
    const float* s = seq + p * 144 + j * 6;
    float d0 = s[0] * inv, d1 = s[1] * inv, d2 = s[2] * inv;
    float d3 = s[3] * inv, d4 = s[4] * inv, d5 = s[5] * inv;
    // Gram-Schmidt
    float n1  = fmaxf(sqrtf(d0 * d0 + d1 * d1 + d2 * d2), 1e-8f);
    float b1x = d0 / n1, b1y = d1 / n1, b1z = d2 / n1;
    float dt  = b1x * d3 + b1y * d4 + b1z * d5;
    float px  = d3 - dt * b1x, py = d4 - dt * b1y, pz = d5 - dt * b1z;
    float n2  = fmaxf(sqrtf(px * px + py * py + pz * pz), 1e-8f);
    float b2x = px / n2, b2y = py / n2, b2z = pz / n2;
    float b3x = b1y * b2z - b1z * b2y;
    float b3y = b1z * b2x - b1x * b2z;
    float b3z = b1x * b2y - b1y * b2x;
    // matrix -> axis-angle (rows are b1,b2,b3)
    float tr  = b1x + b2y + b3z;
    float cs  = fminf(fmaxf((tr - 1.f) * 0.5f, -1.f + 1e-6f), 1.f - 1e-6f);
    float ang = acosf(cs);
    float sn  = sqrtf(fmaxf(1.f - cs * cs, 1e-12f));
    float vx  = b3y - b2z, vy = b1z - b3x, vz = b2x - b1y;
    float k   = ang / (2.f * sn);
    out[p * 72 + j * 3 + 0] = vx * k;
    out[p * 72 + j * 3 + 1] = vy * k;
    out[p * 72 + j * 3 + 2] = vz * k;
}

// ---------------------------------------------------------------------------
// Launcher
// ---------------------------------------------------------------------------
extern "C" void kernel_launch(void* const* d_in, const int* in_sizes, int n_in,
                              void* d_out, int out_size, void* d_ws, size_t ws_size,
                              hipStream_t stream) {
    const float* data_pred = (const float*)d_in[0];   // (1024,72)
    const float* w_in  = (const float*)d_in[1];       // (512,32)
    const float* b_in  = (const float*)d_in[2];       // (512,)
    const float* w1a   = (const float*)d_in[3];       // (256,512)
    const float* b1a   = (const float*)d_in[4];
    const float* w1b   = (const float*)d_in[5];       // (512,256)
    const float* b1b   = (const float*)d_in[6];
    const float* w2a   = (const float*)d_in[7];       // (256,512)
    const float* b2a   = (const float*)d_in[8];
    const float* w2b   = (const float*)d_in[9];       // (512,256)
    const float* b2b   = (const float*)d_in[10];
    const float* w_out = (const float*)d_in[11];      // (32,512)
    const float* b_out = (const float*)d_in[12];
    float* out = (float*)d_out;                       // (1024,72)

    char* ws = (char*)d_ws;
    float*          seq     = (float*)(ws + 0);                 // 1024*144 f32
    unsigned short* d6T     = (unsigned short*)(ws +  589824);  // 144*1024 bf16
    unsigned short* w_in_b  = (unsigned short*)(ws +  884736);  // 512*32
    unsigned short* w1a_b   = (unsigned short*)(ws +  917504);  // 256*512
    unsigned short* w1b_b   = (unsigned short*)(ws + 1179648);  // 512*256
    unsigned short* w2a_b   = (unsigned short*)(ws + 1441792);
    unsigned short* w2b_b   = (unsigned short*)(ws + 1703936);
    unsigned short* w_out_b = (unsigned short*)(ws + 1966080);  // 32*512

    hipMemsetAsync(seq, 0, 1024 * 144 * sizeof(float), stream);

    // weight conversion to bf16
    cvt_bf16_kernel<<<(512 * 32 + 255) / 256, 256, 0, stream>>>(w_in,  w_in_b,  512 * 32);
    cvt_bf16_kernel<<<(256 * 512 + 255) / 256, 256, 0, stream>>>(w1a,  w1a_b,  256 * 512);
    cvt_bf16_kernel<<<(512 * 256 + 255) / 256, 256, 0, stream>>>(w1b,  w1b_b,  512 * 256);
    cvt_bf16_kernel<<<(256 * 512 + 255) / 256, 256, 0, stream>>>(w2a,  w2a_b,  256 * 512);
    cvt_bf16_kernel<<<(512 * 256 + 255) / 256, 256, 0, stream>>>(w2b,  w2b_b,  512 * 256);
    cvt_bf16_kernel<<<(32 * 512 + 255) / 256, 256, 0, stream>>>(w_out, w_out_b, 32 * 512);

    // axis-angle -> 6D (transposed bf16)
    aa_to_d6_kernel<<<(1024 * 24 + 255) / 256, 256, 0, stream>>>(data_pred, d6T, 1024 * 24);

    // fused MLP: 993 windows * 9 channel-blocks = 8937 fragments, 2 waves/block
    const int nFrag = 993 * 9;
    smoothnet_main<<<(nFrag + 1) / 2, 64, 0, stream>>>(
        d6T, w_in_b, b_in, w1a_b, b1a, w1b_b, b1b,
        w2a_b, b2a, w2b_b, b2b, w_out_b, b_out, seq, nFrag);

    // average + 6D -> axis-angle
    finalize_kernel<<<(1024 * 24 + 255) / 256, 256, 0, stream>>>(seq, out, 1024 * 24);
}
